// EMDLoss_38130719654474
// MI455X (gfx1250) — matrix operand order repaired and checked by reading, not compiled, not run on hardware
//
#include <hip/hip_runtime.h>
#include <math.h>
#include <stdint.h>

// ---------------------------------------------------------------------------
// EMD / Sinkhorn loss for MI455X (gfx1250, wave32, WMMA, async-LDS copies)
//
//   k_init      : zero reduction buffers / flag / output
//   k_norm      : L2-normalize rows of a,b; emit f16 hi/lo split operands
//   k_cost      : C = 1 - A_n B_n^T. Workgroup 64x128 tile; A/B panels staged
//                 to LDS via global_load_async_to_lds_b128 (ASYNCcnt), WMMA
//                 v_wmma_f32_16x16x32_f16 fed from ds_load_b128. 3-term f16
//                 hi/lo split => ~22-bit effective precision. Fused row/col
//                 sums of S for the Sinkhorn marginals.
//   k_weights   : mu = l1(clip(l1(mean_row S))), nu likewise; u=v=0
//   10x { k_u, k_v, k_err } : log-domain Sinkhorn, device-side early stop
//   k_final     : out[b] = sum_ij exp((u_i+v_j-C_ij)/eps) * C_ij
//
// C (134 MB) is materialized in workspace and stays L2-resident (192 MB L2),
// so the ~21 streaming passes of the Sinkhorn phase hit L2, not HBM.
// ---------------------------------------------------------------------------

typedef _Float16 v16h __attribute__((ext_vector_type(16)));
typedef float    v8f  __attribute__((ext_vector_type(8)));

static constexpr int   N_ = 8, P_ = 2048, D_ = 64;
static constexpr float EPS_     = 1e-3f;
static constexpr float INV_EPS_ = 1.0f / EPS_;
static constexpr float THRESH_  = 0.1f;
static constexpr int   MAX_ITER_ = 10;

// workspace layout (bytes)
static constexpr size_t OFF_C    = 0;                                   // N*P*P f32
static constexpr size_t SZ_C     = (size_t)N_ * P_ * P_ * sizeof(float);
static constexpr size_t SZ_HALF  = (size_t)N_ * P_ * D_ * sizeof(_Float16);
static constexpr size_t OFF_AH   = OFF_C  + SZ_C;
static constexpr size_t OFF_AL   = OFF_AH + SZ_HALF;
static constexpr size_t OFF_BH   = OFF_AL + SZ_HALF;
static constexpr size_t OFF_BL   = OFF_BH + SZ_HALF;
static constexpr size_t SZ_VEC   = (size_t)N_ * P_ * sizeof(float);
static constexpr size_t OFF_U    = OFF_BL + SZ_HALF;
static constexpr size_t OFF_V    = OFF_U  + SZ_VEC;
static constexpr size_t OFF_UP   = OFF_V  + SZ_VEC;
static constexpr size_t OFF_MU   = OFF_UP + SZ_VEC;
static constexpr size_t OFF_NU   = OFF_MU + SZ_VEC;
static constexpr size_t OFF_RS   = OFF_NU + SZ_VEC;
static constexpr size_t OFF_CS   = OFF_RS + SZ_VEC;
static constexpr size_t OFF_FLAG = OFF_CS + SZ_VEC;
static constexpr size_t WS_NEED  = OFF_FLAG + 256;

// ---------------------------------------------------------------------------
// gfx1250 async LDS copy (ASYNCcnt-tracked); 16 B per lane per issue.
// vdst = LDS byte offset (AS3 offset == low 32 bits of generic shared ptr).
__device__ __forceinline__ void async_copy_b128(void* lds_ptr, const void* gptr) {
#if defined(__gfx1250__)
    uint32_t laddr = (uint32_t)(uintptr_t)lds_ptr;
    uint64_t gaddr = (uint64_t)(uintptr_t)gptr;
    asm volatile("global_load_async_to_lds_b128 %0, %1, off"
                 :: "v"(laddr), "v"(gaddr) : "memory");
#else
    (void)lds_ptr; (void)gptr;
#endif
}
__device__ __forceinline__ void wait_async_all() {
#if defined(__gfx1250__)
    asm volatile("s_wait_asynccnt 0" ::: "memory");
#endif
}

__device__ __forceinline__ float block_reduce_sum(float val, float* sh) {
    int tid = threadIdx.x;
    sh[tid] = val;
    __syncthreads();
    for (int off = 128; off > 0; off >>= 1) {
        if (tid < off) sh[tid] += sh[tid + off];
        __syncthreads();
    }
    float r = sh[0];
    __syncthreads();
    return r;
}

// ---------------------------------------------------------------------------
__global__ __launch_bounds__(256) void k_init(float* rowsum, float* colsum,
                                              float* out, int* flag) {
    int tid = blockIdx.x * blockDim.x + threadIdx.x;
    int stride = gridDim.x * blockDim.x;
    for (int i = tid; i < N_ * P_; i += stride) { rowsum[i] = 0.f; colsum[i] = 0.f; }
    if (tid < N_) out[tid] = 0.f;
    if (tid == 0) *flag = 0;
}

// ---------------------------------------------------------------------------
// One wave per 64-element row: L2-normalize and emit f16 hi/lo split.
__global__ __launch_bounds__(256) void k_norm(const float* __restrict__ a,
                                              const float* __restrict__ b,
                                              _Float16* __restrict__ ah,
                                              _Float16* __restrict__ al,
                                              _Float16* __restrict__ bh,
                                              _Float16* __restrict__ bl) {
    int wid  = blockIdx.x * (blockDim.x >> 5) + (threadIdx.x >> 5);
    int lane = threadIdx.x & 31;
    const int rows = N_ * P_;
    if (wid >= 2 * rows) return;
    const float* src; _Float16 *dh, *dl; int row;
    if (wid < rows) { src = a; dh = ah; dl = al; row = wid; }
    else            { src = b; dh = bh; dl = bl; row = wid - rows; }

    float2 x = *(const float2*)(src + (size_t)row * D_ + lane * 2);
    float ss = x.x * x.x + x.y * x.y;
    ss += __shfl_xor(ss, 1, 32);
    ss += __shfl_xor(ss, 2, 32);
    ss += __shfl_xor(ss, 4, 32);
    ss += __shfl_xor(ss, 8, 32);
    ss += __shfl_xor(ss, 16, 32);
    float scale = 1.0f / fmaxf(sqrtf(ss), 1e-12f);
    float y0 = x.x * scale, y1 = x.y * scale;
    _Float16 h0 = (_Float16)y0, h1 = (_Float16)y1;
    _Float16 l0 = (_Float16)(y0 - (float)h0), l1 = (_Float16)(y1 - (float)h1);
    size_t o = (size_t)row * D_ + lane * 2;
    dh[o] = h0; dh[o + 1] = h1;
    dl[o] = l0; dl[o + 1] = l1;
}

// ---------------------------------------------------------------------------
// Workgroup computes a 64(M) x 128(N) tile of C[b] = 1 - A_n B_n^T.
// 8 waves; wave w owns M-strip (w&3)*16, N-half (w>>2)*64 -> 4 WMMA tiles.
// Panels staged to LDS with async copies; fragments per ISA 7.12.2 layouts.
static constexpr int BM_ = 64, BN_ = 128;
static constexpr int A_CH = (BM_ * D_ * 2) / 16;         // 512 16B chunks per A array
static constexpr int B_CH = (BN_ * D_ * 2) / 16;         // 1024 per B array
static constexpr int TOT_CH = 2 * A_CH + 2 * B_CH;       // 3072

__global__ __launch_bounds__(256) void k_cost(const _Float16* __restrict__ ah,
                                              const _Float16* __restrict__ al,
                                              const _Float16* __restrict__ bh,
                                              const _Float16* __restrict__ bl,
                                              float* __restrict__ C,
                                              float* __restrict__ rowsum,
                                              float* __restrict__ colsum) {
    __shared__ _Float16 sAh[BM_ * D_];   //  8 KB
    __shared__ _Float16 sAl[BM_ * D_];   //  8 KB
    __shared__ _Float16 sBh[BN_ * D_];   // 16 KB
    __shared__ _Float16 sBl[BN_ * D_];   // 16 KB

    const int b   = blockIdx.z;
    const int mi0 = blockIdx.x * BM_;
    const int nj0 = blockIdx.y * BN_;
    const int tid = threadIdx.x;

    const _Float16* Ah = ah + (size_t)b * P_ * D_;
    const _Float16* Al = al + (size_t)b * P_ * D_;
    const _Float16* Bh = bh + (size_t)b * P_ * D_;
    const _Float16* Bl = bl + (size_t)b * P_ * D_;

    // ---- stage A/B panels into LDS (async, 16 B per lane per issue) ----
    #pragma unroll
    for (int it = 0; it < TOT_CH / 256; ++it) {
        int c = tid + it * 256;
        if (c < A_CH) {                               // A hi
            int r = c >> 3, cc = (c & 7) * 8;
            async_copy_b128(&sAh[r * D_ + cc], Ah + (size_t)(mi0 + r) * D_ + cc);
        } else if (c < 2 * A_CH) {                    // A lo
            int k = c - A_CH; int r = k >> 3, cc = (k & 7) * 8;
            async_copy_b128(&sAl[r * D_ + cc], Al + (size_t)(mi0 + r) * D_ + cc);
        } else if (c < 2 * A_CH + B_CH) {             // B hi
            int k = c - 2 * A_CH; int r = k >> 3, cc = (k & 7) * 8;
            async_copy_b128(&sBh[r * D_ + cc], Bh + (size_t)(nj0 + r) * D_ + cc);
        } else {                                      // B lo
            int k = c - 2 * A_CH - B_CH; int r = k >> 3, cc = (k & 7) * 8;
            async_copy_b128(&sBl[r * D_ + cc], Bl + (size_t)(nj0 + r) * D_ + cc);
        }
    }
    wait_async_all();       // this wave's async copies complete
    __syncthreads();        // all waves' copies visible

    const int w    = tid >> 5;
    const int lane = tid & 31;
    const int half = lane >> 4;
    const int l15  = lane & 15;
    const int mloc = (w & 3) * 16;          // M-strip within panel
    const int nbas = (w >> 2) * 64;         // N-half within panel

    // ---- A fragments (reused across the 4 N-tiles), from LDS ----
    const int arow = mloc + l15;
    v16h a_h[2], a_l[2];
    #pragma unroll
    for (int kk = 0; kk < 2; ++kk) {
        const int ka = kk * 32 + half * 8;
        *(uint4*)&a_h[kk]       = *(const uint4*)(sAh + arow * D_ + ka);
        *((uint4*)&a_h[kk] + 1) = *(const uint4*)(sAh + arow * D_ + ka + 16);
        *(uint4*)&a_l[kk]       = *(const uint4*)(sAl + arow * D_ + ka);
        *((uint4*)&a_l[kk] + 1) = *(const uint4*)(sAl + arow * D_ + ka + 16);
    }

    float* Cb = C + (size_t)b * P_ * P_;

    #pragma unroll
    for (int t = 0; t < 4; ++t) {
        const int nloc = nbas + t * 16;
        const int brow = nloc + l15;
        v8f acc = {0.f, 0.f, 0.f, 0.f, 0.f, 0.f, 0.f, 0.f};
        #pragma unroll
        for (int kk = 0; kk < 2; ++kk) {
            const int kc = kk * 32 + half * 8;
            v16h b_h, b_l;
            *(uint4*)&b_h       = *(const uint4*)(sBh + brow * D_ + kc);
            *((uint4*)&b_h + 1) = *(const uint4*)(sBh + brow * D_ + kc + 16);
            *(uint4*)&b_l       = *(const uint4*)(sBl + brow * D_ + kc);
            *((uint4*)&b_l + 1) = *(const uint4*)(sBl + brow * D_ + kc + 16);
            // (Ah+Al)(Bh+Bl) ~= Ah*Bh + Ah*Bl + Al*Bh
            acc = __builtin_amdgcn_wmma_f32_16x16x32_f16(false, a_h[kk], false, b_h,
                                                         (short)0, acc, false, false);
            acc = __builtin_amdgcn_wmma_f32_16x16x32_f16(false, a_h[kk], false, b_l,
                                                         (short)0, acc, false, false);
            acc = __builtin_amdgcn_wmma_f32_16x16x32_f16(false, a_l[kk], false, b_h,
                                                         (short)0, acc, false, false);
        }

        // store C = 1 - S; C layout: VGPR r, lane l -> M=r+8*half, N=l&15
        const int mi = mi0 + mloc, nj = nj0 + nloc;
        float scol = 0.f;
        #pragma unroll
        for (int r = 0; r < 8; ++r) {
            int M = mi + r + half * 8;
            Cb[(size_t)M * P_ + nj + l15] = 1.0f - acc[r];
            scol += acc[r];
        }
        scol += __shfl_xor(scol, 16, 32);
        if (half == 0) atomicAdd(&colsum[b * P_ + nj + l15], scol);
        #pragma unroll
        for (int r = 0; r < 8; ++r) {
            float v = acc[r];
            v += __shfl_xor(v, 1, 32);
            v += __shfl_xor(v, 2, 32);
            v += __shfl_xor(v, 4, 32);
            v += __shfl_xor(v, 8, 32);
            if (l15 == r) atomicAdd(&rowsum[b * P_ + mi + r + half * 8], v);
        }
    }
}

// ---------------------------------------------------------------------------
__global__ __launch_bounds__(256) void k_weights(const float* __restrict__ rowsum,
                                                 const float* __restrict__ colsum,
                                                 float* __restrict__ mu,
                                                 float* __restrict__ nu,
                                                 float* __restrict__ u,
                                                 float* __restrict__ v,
                                                 float* __restrict__ uprev) {
    __shared__ float sh[256];
    int b = blockIdx.x, tid = threadIdx.x;
    const float invP = 1.0f / (float)P_;

    {   // rows -> mu; also zero u, uprev
        const float* rs = rowsum + b * P_;
        float s = 0.f;
        for (int i = tid; i < P_; i += 256) s += fabsf(rs[i] * invP);
        float l1 = fmaxf(block_reduce_sum(s, sh), 1e-12f);
        float s2 = 0.f;
        for (int i = tid; i < P_; i += 256) s2 += fmaxf(rs[i] * invP / l1, 0.f);
        float l1b = fmaxf(block_reduce_sum(s2, sh), 1e-12f);
        for (int i = tid; i < P_; i += 256) {
            mu[b * P_ + i] = fmaxf(rs[i] * invP / l1, 0.f) / l1b;
            u[b * P_ + i] = 0.f;
            uprev[b * P_ + i] = 0.f;
        }
    }
    __syncthreads();
    {   // cols -> nu; also zero v
        const float* cs = colsum + b * P_;
        float s = 0.f;
        for (int i = tid; i < P_; i += 256) s += fabsf(cs[i] * invP);
        float l1 = fmaxf(block_reduce_sum(s, sh), 1e-12f);
        float s2 = 0.f;
        for (int i = tid; i < P_; i += 256) s2 += fmaxf(cs[i] * invP / l1, 0.f);
        float l1b = fmaxf(block_reduce_sum(s2, sh), 1e-12f);
        for (int i = tid; i < P_; i += 256) {
            nu[b * P_ + i] = fmaxf(cs[i] * invP / l1, 0.f) / l1b;
            v[b * P_ + i] = 0.f;
        }
    }
}

// ---------------------------------------------------------------------------
// u_i = eps*(log(mu_i+1e-8) - LSE_j((v_j - C_ij)/eps))   (u_i/eps cancels)
__global__ __launch_bounds__(256) void k_u(const float* __restrict__ C,
                                           const float* __restrict__ vvec,
                                           const float* __restrict__ mu,
                                           float* __restrict__ u,
                                           float* __restrict__ uprev,
                                           const int* __restrict__ flag) {
    if (*flag) return;   // early-stopped
    int i = blockIdx.x, b = blockIdx.y, tid = threadIdx.x;
    const float* Crow = C + ((size_t)b * P_ + i) * P_;
    const float* vb   = vvec + b * P_;
    float m = -INFINITY, s = 0.f;
    for (int j = tid; j < P_; j += 256) {
        __builtin_prefetch(Crow + j + 1024, 0, 1);
        float x = (vb[j] - Crow[j]) * INV_EPS_;
        if (x > m) { s = s * __expf(m - x) + 1.0f; m = x; }
        else       { s += __expf(x - m); }
    }
    __shared__ float sm[256], ss[256];
    sm[tid] = m; ss[tid] = s;
    __syncthreads();
    for (int off = 128; off > 0; off >>= 1) {
        if (tid < off) {
            float m2 = sm[tid + off], s2 = ss[tid + off];
            float M = fmaxf(sm[tid], m2);
            ss[tid] = ss[tid] * __expf(sm[tid] - M) + s2 * __expf(m2 - M);
            sm[tid] = M;
        }
        __syncthreads();
    }
    if (tid == 0) {
        float lse = sm[0] + __logf(ss[0]);
        uprev[b * P_ + i] = u[b * P_ + i];
        u[b * P_ + i] = EPS_ * (__logf(mu[b * P_ + i] + 1e-8f) - lse);
    }
}

// ---------------------------------------------------------------------------
// v_j = eps*(log(nu_j+1e-8) - LSE_i((u_i - C_ij)/eps))   (v_j/eps cancels)
__global__ __launch_bounds__(256) void k_v(const float* __restrict__ C,
                                           const float* __restrict__ uvec,
                                           const float* __restrict__ nu,
                                           float* __restrict__ v,
                                           const int* __restrict__ flag) {
    if (*flag) return;
    int cg = blockIdx.x, b = blockIdx.y, tid = threadIdx.x;
    int j  = cg * 32 + (tid & 31);
    int r0 = tid >> 5;                 // 0..7
    const float* Cb = C + (size_t)b * P_ * P_;
    const float* ub = uvec + b * P_;
    float m = -INFINITY, s = 0.f;
    for (int i = r0; i < P_; i += 8) {
        __builtin_prefetch(Cb + (size_t)(i + 32) * P_ + j, 0, 1);
        float x = (ub[i] - Cb[(size_t)i * P_ + j]) * INV_EPS_;
        if (x > m) { s = s * __expf(m - x) + 1.0f; m = x; }
        else       { s += __expf(x - m); }
    }
    __shared__ float sm[256], ss[256];
    sm[tid] = m; ss[tid] = s;
    __syncthreads();
    if (r0 == 0) {
        float M = sm[tid], S = ss[tid];
        #pragma unroll
        for (int rr = 1; rr < 8; ++rr) {
            float m2 = sm[tid + rr * 32], s2 = ss[tid + rr * 32];
            float Mn = fmaxf(M, m2);
            S = S * __expf(M - Mn) + s2 * __expf(m2 - Mn);
            M = Mn;
        }
        float lse = M + __logf(S);
        v[b * P_ + j] = EPS_ * (__logf(nu[b * P_ + j] + 1e-8f) - lse);
    }
}

// ---------------------------------------------------------------------------
__global__ __launch_bounds__(256) void k_err(const float* __restrict__ u,
                                             const float* __restrict__ uprev,
                                             int* __restrict__ flag) {
    if (*flag) return;
    __shared__ float sh[256];
    int tid = threadIdx.x;
    float s = 0.f;
    for (int i = tid; i < N_ * P_; i += 256) s += fabsf(u[i] - uprev[i]);
    float tot = block_reduce_sum(s, sh);
    if (tid == 0) {
        float err = tot / (float)N_;   // mean over batch of per-batch sums
        if (err < THRESH_) *flag = 1;
    }
}

// ---------------------------------------------------------------------------
__global__ __launch_bounds__(256) void k_final(const float* __restrict__ C,
                                               const float* __restrict__ u,
                                               const float* __restrict__ v,
                                               float* __restrict__ out) {
    int i = blockIdx.x, b = blockIdx.y, tid = threadIdx.x;
    const float* Crow = C + ((size_t)b * P_ + i) * P_;
    const float* vb = v + b * P_;
    float ui = u[b * P_ + i];
    float s = 0.f;
    for (int j = tid; j < P_; j += 256) {
        __builtin_prefetch(Crow + j + 1024, 0, 1);
        float c = Crow[j];
        s += __expf((ui + vb[j] - c) * INV_EPS_) * c;
    }
    __shared__ float sh[256];
    float tot = block_reduce_sum(s, sh);
    if (tid == 0) atomicAdd(&out[b], tot);
}

// ---------------------------------------------------------------------------
extern "C" void kernel_launch(void* const* d_in, const int* in_sizes, int n_in,
                              void* d_out, int out_size, void* d_ws, size_t ws_size,
                              hipStream_t stream) {
    (void)in_sizes; (void)n_in; (void)out_size;
    if (ws_size < WS_NEED) return;

    const float* a = (const float*)d_in[0];
    const float* b = (const float*)d_in[1];
    float* out = (float*)d_out;

    char* ws = (char*)d_ws;
    float*    C      = (float*)(ws + OFF_C);
    _Float16* ah     = (_Float16*)(ws + OFF_AH);
    _Float16* al     = (_Float16*)(ws + OFF_AL);
    _Float16* bh     = (_Float16*)(ws + OFF_BH);
    _Float16* bl     = (_Float16*)(ws + OFF_BL);
    float*    u      = (float*)(ws + OFF_U);
    float*    v      = (float*)(ws + OFF_V);
    float*    uprev  = (float*)(ws + OFF_UP);
    float*    mu     = (float*)(ws + OFF_MU);
    float*    nu     = (float*)(ws + OFF_NU);
    float*    rowsum = (float*)(ws + OFF_RS);
    float*    colsum = (float*)(ws + OFF_CS);
    int*      flag   = (int*)(ws + OFF_FLAG);

    k_init<<<64, 256, 0, stream>>>(rowsum, colsum, out, flag);

    {   // 2*N*P rows, 8 waves (rows) per 256-thread block
        int rows = 2 * N_ * P_;
        k_norm<<<(rows + 7) / 8, 256, 0, stream>>>(a, b, ah, al, bh, bl);
    }

    {   // workgroup = 64x128 tile of C
        dim3 grid(P_ / BM_, P_ / BN_, N_);
        k_cost<<<grid, 256, 0, stream>>>(ah, al, bh, bl, C, rowsum, colsum);
    }

    k_weights<<<N_, 256, 0, stream>>>(rowsum, colsum, mu, nu, u, v, uprev);

    for (int it = 0; it < MAX_ITER_; ++it) {
        k_u<<<dim3(P_, N_), 256, 0, stream>>>(C, v, mu, u, uprev, flag);
        k_v<<<dim3(P_ / 32, N_), 256, 0, stream>>>(C, u, nu, v, flag);
        k_err<<<1, 256, 0, stream>>>(u, uprev, flag);
    }

    k_final<<<dim3(P_, N_), 256, 0, stream>>>(C, u, v, out);
}
